// FeatureExtractor_48490180772408
// MI455X (gfx1250) — compile-verified, matrix operand order checked by reading, not run on hardware
//
#include <hip/hip_runtime.h>
#include <hip/hip_bf16.h>
#include <cstdint>

#define H_ 128
#define W_ 128
#define S_ 16384            // H_*W_
#define PW 130              // padded row width (1-pixel halo)
#define PS (130*130)        // padded pixels per image
#define BATCH 8
#define TB 16               // T*B
#define THRESH_ 0.5f
#define DECAY_ 0.25f

typedef __bf16 bf16_t;
typedef __attribute__((ext_vector_type(16))) __bf16 v16bf;
typedef __attribute__((ext_vector_type(8)))  __bf16 v8bf;
typedef __attribute__((ext_vector_type(8)))  float  v8f;

__device__ __forceinline__ float bf2f(bf16_t b) {
  unsigned short u = __builtin_bit_cast(unsigned short, b);
  unsigned int x = ((unsigned int)u) << 16;
  return __builtin_bit_cast(float, x);
}
__device__ __forceinline__ bf16_t f2bf(float f) {
  unsigned int u = __builtin_bit_cast(unsigned int, f);
  unsigned int r = (u + 0x7FFFu + ((u >> 16) & 1u)) >> 16;
  return __builtin_bit_cast(bf16_t, (unsigned short)r);
}
// mem_update over T=2 (forward pass)
__device__ __forceinline__ void lif2(float y0, float y1, float& s0, float& s1) {
  s0 = (y0 > THRESH_) ? 1.f : 0.f;
  float mem1 = y0 * DECAY_ * (1.f - s0) + y1;
  s1 = (mem1 > THRESH_) ? 1.f : 0.f;
}
// padded NHWC element index for (image n, unpadded pixel, channel c)
__device__ __forceinline__ long pidx(int n, int pix, int c) {
  return ((((long)n * PS) + (pix >> 7) * PW + (pix & 127) + PW + 1) << 6) + c;
}

// ---------------------------------------------------------------------------
// WMMA implicit-GEMM conv 64->64, 3x3 SAME. Input: halo-padded NHWC bf16
// (16, PS, 64); output raw f32 NHWC (16, S_, 64). Block = 8 waves; tile =
// 64 Cout x 128 pixels (one output row). K = 576 in 18 chunks of 32 (tap-
// major). All fragment loads are immediate offsets off one per-lane base:
// no bounds branches, no EXEC juggling in the unrolled hot loop.
// ---------------------------------------------------------------------------
__global__ __launch_bounds__(256)
void conv64_wmma_kernel(const bf16_t* __restrict__ in,
                        const bf16_t* __restrict__ wp,   // packed (18,4,32,16)
                        float* __restrict__ raw)
{
  const int blk  = blockIdx.x;      // n*128 + h
  const int n    = blk >> 7;
  const int h    = blk & 127;
  const int tid  = threadIdx.x;
  const int wave = tid >> 5;
  const int lane = tid & 31;
  const int m     = wave & 3;           // Cout tile (16 rows)
  const int px0   = (wave >> 2) << 6;   // pixel base: 0 or 64
  const int lcol  = lane & 15;
  const int lhigh = lane >> 4;
  const int kbase = lhigh << 3;         // per-lane K sub-base (ISA 16-bit layout)

  v8f acc[4];
#pragma unroll
  for (int s = 0; s < 4; ++s)
#pragma unroll
    for (int r = 0; r < 8; ++r) acc[s][r] = 0.f;

  // base: padded position of (h-1, px0+lcol-1), channel kbase
  const bf16_t* bbase = in + (((long)n * PS + (long)h * PW + px0 + lcol) << 6) + kbase;
  const bf16_t* abase = wp + ((m * 32 + lane) << 4);

#pragma unroll
  for (int kc = 0; kc < 18; ++kc) {
    const int tap  = kc >> 1;
    const int kh   = tap / 3;
    const int kw   = tap - kh * 3;
    const int coff = (kc & 1) << 5;     // channel half 0/32
    v16bf afrag = *(const v16bf*)(abase + (kc << 11));   // +kc*4096B
#pragma unroll
    for (int s = 0; s < 4; ++s) {
      const bf16_t* src = bbase + (long)((kh * PW + kw) << 6) + (s << 10) + coff;
      v8bf lo = *(const v8bf*)(src);        // K = kbase..kbase+7
      v8bf hi = *(const v8bf*)(src + 16);   // K = 16+kbase..16+kbase+7
      v16bf bfrag;
#pragma unroll
      for (int j = 0; j < 8; ++j) { bfrag[j] = lo[j]; bfrag[j + 8] = hi[j]; }
      acc[s] = __builtin_amdgcn_wmma_f32_16x16x32_bf16(
          false, afrag, false, bfrag, (short)0, acc[s], false, false);
    }
  }

  const int co0 = (m << 4) + (lhigh << 3);
  const long outBase = (long)n * S_ * 64;
#pragma unroll
  for (int s = 0; s < 4; ++s) {
    const int px = px0 + (s << 4) + lcol;
    *(v8f*)(raw + outBase + ((long)(h * W_ + px) << 6) + co0) = acc[s];
  }
}

// Pack OIHW f32 (64,64,3,3) -> fragment-major bf16 per ISA A-layout.
__global__ void pack_w64_kernel(const float* __restrict__ w, bf16_t* __restrict__ out)
{
  int idx = blockIdx.x * 256 + threadIdx.x;   // 64*576 = 36864
  if (idx >= 64 * 576) return;
  int j    = idx & 15;
  int lane = (idx >> 4) & 31;
  int m    = (idx >> 9) & 3;
  int kc   = idx >> 11;
  int co   = (m << 4) + (lane & 15);
  int kb   = (lane >> 4) << 3;
  int kk   = (j < 8) ? (kb + j) : (16 + kb + (j - 8));
  int ci   = ((kc & 1) << 5) + kk;
  int tap  = kc >> 1;
  out[idx] = f2bf(w[((co * 64 + ci) * 9) + tap]);
}

// ---------------------------------------------------------------------------
// Direct f32 conv 3x3 SAME (small layers, <5% of FLOPs).
// ---------------------------------------------------------------------------
__global__ void conv_direct_kernel(const float* __restrict__ in, const float* __restrict__ w,
                                   const float* __restrict__ bias, float* __restrict__ out,
                                   int N, int CinTot, int c0, int Cin,
                                   int CoutTot, int co0, int Cout)
{
  long idx = (long)blockIdx.x * 256 + threadIdx.x;
  long total = (long)N * Cout * S_;
  if (idx >= total) return;
  int pix = (int)(idx & (S_ - 1));
  int t   = (int)(idx >> 14);
  int co  = t % Cout;
  int n   = t / Cout;
  int h = pix >> 7, x = pix & 127;
  float acc = bias ? bias[co] : 0.f;
  const float* wrow = w + (long)co * Cin * 9;
  const float* ip   = in + ((long)n * CinTot + c0) * S_;
  for (int ci = 0; ci < Cin; ++ci) {
    const float* ic = ip + (long)ci * S_;
    const float* wc = wrow + ci * 9;
#pragma unroll
    for (int kh = 0; kh < 3; ++kh) {
      int hh = h + kh - 1;
      if ((unsigned)hh >= (unsigned)H_) continue;
#pragma unroll
      for (int kw = 0; kw < 3; ++kw) {
        int ww = x + kw - 1;
        if ((unsigned)ww >= (unsigned)W_) continue;
        acc += ic[hh * W_ + ww] * wc[kh * 3 + kw];
      }
    }
  }
  out[((long)n * CoutTot + co0 + co) * S_ + pix] = acc;
}

// -------------------------- TDBN statistics -------------------------------
__global__ void zero_kernel(float* p, int n) {
  int i = blockIdx.x * 256 + threadIdx.x;
  if (i < n) p[i] = 0.f;
}
__global__ void zero16_kernel(uint4* p, long n16) {
  long i = (long)blockIdx.x * 256 + threadIdx.x;
  if (i < n16) p[i] = make_uint4(0u, 0u, 0u, 0u);
}

// NCHW stats: grid(C, N); sums[c]=sum, sums[C+c]=sumsq
__global__ void stats_nchw_kernel(const float* __restrict__ x, float* __restrict__ sums, int C)
{
  __shared__ float ls[256], lq[256];
  const int c = blockIdx.x, n = blockIdx.y;
  const float* p = x + ((long)n * C + c) * S_;
  float s = 0.f, q = 0.f;
  for (int i = threadIdx.x; i < S_; i += 256) { float v = p[i]; s += v; q += v * v; }
  ls[threadIdx.x] = s; lq[threadIdx.x] = q;
  __syncthreads();
  for (int off = 128; off > 0; off >>= 1) {
    if (threadIdx.x < off) {
      ls[threadIdx.x] += ls[threadIdx.x + off];
      lq[threadIdx.x] += lq[threadIdx.x + off];
    }
    __syncthreads();
  }
  if (threadIdx.x == 0) { atomicAdd(&sums[c], ls[0]); atomicAdd(&sums[C + c], lq[0]); }
}

// NHWC stats for C=64: block = 64 channels x 4 pixel lanes
__global__ void stats_nhwc64_kernel(const float* __restrict__ x, float* __restrict__ sums)
{
  __shared__ float ls[256], lq[256];
  const int tid = threadIdx.x;
  const int c   = tid & 63;
  const int pl  = tid >> 6;
  const long totalPix = (long)TB * S_;
  const long ppb = (totalPix + gridDim.x - 1) / gridDim.x;
  long p0 = (long)blockIdx.x * ppb;
  long p1 = p0 + ppb; if (p1 > totalPix) p1 = totalPix;
  float s = 0.f, q = 0.f;
  for (long p = p0 + pl; p < p1; p += 4) { float v = x[p * 64 + c]; s += v; q += v * v; }
  ls[tid] = s; lq[tid] = q;
  __syncthreads();
  if (tid < 64) {
    float S = ls[tid] + ls[tid + 64] + ls[tid + 128] + ls[tid + 192];
    float Q = lq[tid] + lq[tid + 64] + lq[tid + 128] + lq[tid + 192];
    atomicAdd(&sums[c], S);
    atomicAdd(&sums[64 + c], Q);
  }
}

// scale/shift: stats[2C+c]=g*rsqrt(var+eps); stats[3C+c]=be-mean*scale
__global__ void bn_finalize_kernel(float* stats, const float* __restrict__ g,
                                   const float* __restrict__ be, int C, float inv_count)
{
  int c = threadIdx.x;
  if (c >= C) return;
  float mean = stats[c] * inv_count;
  float var  = stats[C + c] * inv_count - mean * mean;
  float sc   = g[c] * rsqrtf(var + 1e-5f);
  stats[2 * C + c] = sc;
  stats[3 * C + c] = be[c] - mean * sc;
}

// -------------------------- BN + LIF variants -----------------------------
// identical timesteps (pre-first-LIF stages): raw (Bc,C,S) -> out (2,Bc,C,S)
__global__ void bnlif_single_nchw_kernel(const float* __restrict__ raw, const float* __restrict__ stats,
                                         float* __restrict__ out, int Bc, int C)
{
  long idx = (long)blockIdx.x * 256 + threadIdx.x;
  long plane = (long)Bc * C * S_;
  if (idx >= plane) return;
  int c = (int)((idx >> 14) % C);
  float y = raw[idx] * stats[2 * C + c] + stats[3 * C + c];
  float s0, s1; lif2(y, y, s0, s1);
  out[idx] = s0;
  out[idx + plane] = s1;
}

// two timesteps, NCHW f32 -> NCHW f32
__global__ void bnlif_pair_nchw_kernel(const float* __restrict__ raw, const float* __restrict__ stats,
                                       float* __restrict__ out, int Bc, int C)
{
  long idx = (long)blockIdx.x * 256 + threadIdx.x;
  long plane = (long)Bc * C * S_;
  if (idx >= plane) return;
  int c = (int)((idx >> 14) % C);
  float sc = stats[2 * C + c], sh = stats[3 * C + c];
  float y0 = raw[idx] * sc + sh;
  float y1 = raw[idx + plane] * sc + sh;
  float s0, s1; lif2(y0, y1, s0, s1);
  out[idx] = s0;
  out[idx + plane] = s1;
}

// conv_main: raw (2,8,64,S) NCHW f32 -> spikes padded NHWC bf16
__global__ void bnlif_pair_nchw_to_nhwc_kernel(const float* __restrict__ raw,
                                               const float* __restrict__ stats,
                                               bf16_t* __restrict__ spk)
{
  long idx = (long)blockIdx.x * 256 + threadIdx.x;
  long plane = (long)BATCH * 64 * S_;
  if (idx >= plane) return;
  int pix = (int)(idx & (S_ - 1));
  int c   = (int)((idx >> 14) & 63);
  int n   = (int)(idx >> 20);
  float sc = stats[128 + c], sh = stats[192 + c];
  float y0 = raw[idx] * sc + sh;
  float y1 = raw[idx + plane] * sc + sh;
  float s0, s1; lif2(y0, y1, s0, s1);
  long o0 = pidx(n, pix, c);
  spk[o0] = f2bf(s0);
  spk[o0 + (long)BATCH * PS * 64] = f2bf(s1);
}

// blocks: raw NHWC f32 (+ optional padded residual spikes) -> padded spikes
__global__ void bnlif_nhwc_kernel(const float* __restrict__ raw, const float* __restrict__ stats,
                                  bf16_t* __restrict__ spk, const bf16_t* __restrict__ res)
{
  long idx = (long)blockIdx.x * 256 + threadIdx.x;   // raw (n,pix,c), n<8 = t0
  long plane = (long)BATCH * S_ * 64;
  if (idx >= plane) return;
  int c   = (int)(idx & 63);
  int pix = (int)((idx >> 6) & (S_ - 1));
  int n   = (int)(idx >> 20);
  float sc = stats[128 + c], sh = stats[192 + c];
  float y0 = raw[idx] * sc + sh;
  float y1 = raw[idx + plane] * sc + sh;
  long o0 = pidx(n, pix, c);
  long planeP = (long)BATCH * PS * 64;
  if (res) { y0 += bf2f(res[o0]); y1 += bf2f(res[o0 + planeP]); }
  float s0, s1; lif2(y0, y1, s0, s1);
  spk[o0] = f2bf(s0);
  spk[o0 + planeP] = f2bf(s1);
}

// -------------------------- misc elementwise ------------------------------
__global__ void tmean_kernel(const float* __restrict__ x, float* __restrict__ o, long plane)
{
  long i = (long)blockIdx.x * 256 + threadIdx.x;
  if (i < plane) o[i] = 0.5f * (x[i] + x[i + plane]);
}
__global__ void mul_kernel(const float* __restrict__ a, const float* __restrict__ b,
                           float* __restrict__ o, long n)
{
  long i = (long)blockIdx.x * 256 + threadIdx.x;
  if (i < n) o[i] = a[i] * b[i];
}
// d_out main: mean_t(cur + tmp), padded NHWC bf16 -> NCHW f32
__global__ void final_main_kernel(const bf16_t* __restrict__ cur, const bf16_t* __restrict__ tmp,
                                  float* __restrict__ out)
{
  long idx = (long)blockIdx.x * 256 + threadIdx.x;
  long total = (long)BATCH * 64 * S_;
  if (idx >= total) return;
  int pix = (int)(idx & (S_ - 1));
  int c   = (int)((idx >> 14) & 63);
  int b   = (int)(idx >> 20);
  long i0 = pidx(b, pix, c);
  long i1 = i0 + (long)BATCH * PS * 64;
  out[idx] = 0.5f * (bf2f(cur[i0]) + bf2f(cur[i1]) + bf2f(tmp[i0]) + bf2f(tmp[i1]));
}
__global__ void final_est_kernel(const float* __restrict__ est, float* __restrict__ out)
{
  long idx = (long)blockIdx.x * 256 + threadIdx.x;
  long plane = (long)BATCH * 4 * S_;
  if (idx >= plane) return;
  out[idx] = 0.5f * (est[idx] + est[idx + plane]);
}

// ===========================================================================
extern "C" void kernel_launch(void* const* d_in, const int* in_sizes, int n_in,
                              void* d_out, int out_size, void* d_ws, size_t ws_size,
                              hipStream_t stream)
{
  (void)in_sizes; (void)n_in; (void)out_size; (void)ws_size;
  // Leaf order (insertion order of setup_inputs dicts):
  // 0:x | 1..8: conv0[i].{w,b} | 9..16: bn0[i].{g,be} | 17..24: conv1[i].{w,b}
  // 25..32: ca.{c1.w,c1.b,bn1.g,bn1.be,c2.w,c2.b,bn2.g,bn2.be}
  // 33,34: conv_main.{w,b} | 35,36: bn_main.{g,be}
  // 37+12k: blocks[k].{c1.w,c1.b,bn1.g,bn1.be,c2.w,c2.b,bn2.g,bn2.be,c3.w,c3.b,bn3.g,bn3.be}
  const float* x = (const float*)d_in[0];
  auto F = [&](int i) { return (const float*)d_in[i]; };

  char* ws = (char*)d_ws;
  size_t off = 0;
  auto alloc = [&](size_t bytes) -> void* {
    off = (off + 255) & ~(size_t)255;
    void* p = ws + off;
    off += bytes;
    return p;
  };

  const size_t spkBytes = (size_t)TB * PS * 64 * 2;             // padded bf16
  float*  raw    = (float*) alloc((size_t)TB * 64 * S_ * 4);    // conv raw
  bf16_t* spkA   = (bf16_t*)alloc(spkBytes);
  bf16_t* spkB   = (bf16_t*)alloc(spkBytes);
  bf16_t* spkC   = (bf16_t*)alloc(spkBytes);
  bf16_t* tmp    = (bf16_t*)alloc(spkBytes);
  bf16_t* wpck   = (bf16_t*)alloc((size_t)42 * 64 * 576 * 2);
  float*  est    = (float*) alloc((size_t)TB * 4 * S_ * 4);
  float*  stats  = (float*) alloc(256 * 4);
  float*  braw   = (float*) alloc((size_t)BATCH * 16 * S_ * 4);
  float*  bspk   = (float*) alloc((size_t)TB * 16 * S_ * 4);
  float*  camean = (float*) alloc((size_t)BATCH * 4 * S_ * 4);
  float*  caraw8 = (float*) alloc((size_t)BATCH * 8 * S_ * 4);
  float*  caspk8 = (float*) alloc((size_t)TB * 8 * S_ * 4);
  float*  caraw4 = (float*) alloc((size_t)TB * 4 * S_ * 4);
  float*  wspk4  = (float*) alloc((size_t)TB * 4 * S_ * 4);
  float*  cmin   = (float*) alloc((size_t)TB * 4 * S_ * 4);

  auto G = [](long nthreads) { return dim3((unsigned)((nthreads + 255) / 256)); };
  const float invTBS = 1.f / (float)((long)TB * S_);
  const float invBS  = 1.f / (float)((long)BATCH * S_);
  const long  n16    = (long)(spkBytes / 16);

  // zero padded spike buffers so halos read as 0 (idempotent per call)
  zero16_kernel<<<G(n16), 256, 0, stream>>>((uint4*)spkA, n16);
  zero16_kernel<<<G(n16), 256, 0, stream>>>((uint4*)spkB, n16);
  zero16_kernel<<<G(n16), 256, 0, stream>>>((uint4*)spkC, n16);

  // --- pack WMMA weights for all 42 block convs (c1/c2/c3 at +0/+4/+8) ---
  for (int bi = 0; bi < 14; ++bi)
    for (int L = 0; L < 3; ++L)
      pack_w64_kernel<<<G(64 * 576), 256, 0, stream>>>(
          F(37 + 12 * bi + 4 * L), wpck + (size_t)(bi * 3 + L) * 64 * 576);

  // --- four branches (timesteps identical pre-LIF: conv once per image) ---
  for (int i = 0; i < 4; ++i) {
    int Cin = 41 - 10 * i, c0 = 5 * i;
    conv_direct_kernel<<<G((long)BATCH * 16 * S_), 256, 0, stream>>>(
        x, F(1 + 2 * i), nullptr, braw, BATCH, 41, c0, Cin, 16, 0, 16);
    zero_kernel<<<1, 32, 0, stream>>>(stats, 32);
    stats_nchw_kernel<<<dim3(16, BATCH), 256, 0, stream>>>(braw, stats, 16);
    bn_finalize_kernel<<<1, 64, 0, stream>>>(stats, F(9 + 2 * i), F(10 + 2 * i), 16, invBS);
    bnlif_single_nchw_kernel<<<G((long)BATCH * 16 * S_), 256, 0, stream>>>(braw, stats, bspk, BATCH, 16);
    conv_direct_kernel<<<G((long)TB * 1 * S_), 256, 0, stream>>>(
        bspk, F(17 + 2 * i), F(18 + 2 * i), est, TB, 16, 0, 16, 4, i, 1);
  }

  // --- CA layer ---
  long p4 = (long)BATCH * 4 * S_;
  tmean_kernel<<<G(p4), 256, 0, stream>>>(est, camean, p4);
  conv_direct_kernel<<<G((long)BATCH * 8 * S_), 256, 0, stream>>>(
      camean, F(25), nullptr, caraw8, BATCH, 4, 0, 4, 8, 0, 8);
  zero_kernel<<<1, 32, 0, stream>>>(stats, 16);
  stats_nchw_kernel<<<dim3(8, BATCH), 256, 0, stream>>>(caraw8, stats, 8);
  bn_finalize_kernel<<<1, 64, 0, stream>>>(stats, F(27), F(28), 8, invBS);
  bnlif_single_nchw_kernel<<<G((long)BATCH * 8 * S_), 256, 0, stream>>>(caraw8, stats, caspk8, BATCH, 8);
  conv_direct_kernel<<<G((long)TB * 4 * S_), 256, 0, stream>>>(
      caspk8, F(29), nullptr, caraw4, TB, 8, 0, 8, 4, 0, 4);
  zero_kernel<<<1, 32, 0, stream>>>(stats, 8);
  stats_nchw_kernel<<<dim3(4, TB), 256, 0, stream>>>(caraw4, stats, 4);
  bn_finalize_kernel<<<1, 64, 0, stream>>>(stats, F(31), F(32), 4, invTBS);
  bnlif_pair_nchw_kernel<<<G((long)BATCH * 4 * S_), 256, 0, stream>>>(caraw4, stats, wspk4, BATCH, 4);
  mul_kernel<<<G((long)TB * 4 * S_), 256, 0, stream>>>(wspk4, est, cmin, (long)TB * 4 * S_);

  // --- conv_main 4->64, BN, LIF -> padded NHWC bf16 spikes; save tmp ---
  conv_direct_kernel<<<G((long)TB * 64 * S_), 256, 0, stream>>>(
      cmin, F(33), nullptr, raw, TB, 4, 0, 4, 64, 0, 64);
  zero_kernel<<<1, 128, 0, stream>>>(stats, 128);
  stats_nchw_kernel<<<dim3(64, TB), 256, 0, stream>>>(raw, stats, 64);
  bn_finalize_kernel<<<1, 64, 0, stream>>>(stats, F(35), F(36), 64, invTBS);
  bnlif_pair_nchw_to_nhwc_kernel<<<G((long)BATCH * 64 * S_), 256, 0, stream>>>(raw, stats, spkA);
  hipMemcpyAsync(tmp, spkA, spkBytes, hipMemcpyDeviceToDevice, stream);

  // --- 14 residual blocks (WMMA path) ---
  bf16_t* cur = spkA; bf16_t* s1 = spkB; bf16_t* s2 = spkC;
  const long NE = (long)BATCH * S_ * 64;
  for (int bi = 0; bi < 14; ++bi) {
    int base = 37 + 12 * bi;
    // layer 1: cur -> s1
    conv64_wmma_kernel<<<dim3(TB * 128), 256, 0, stream>>>(cur, wpck + (size_t)(bi * 3 + 0) * 36864, raw);
    zero_kernel<<<1, 128, 0, stream>>>(stats, 128);
    stats_nhwc64_kernel<<<256, 256, 0, stream>>>(raw, stats);
    bn_finalize_kernel<<<1, 64, 0, stream>>>(stats, F(base + 2), F(base + 3), 64, invTBS);
    bnlif_nhwc_kernel<<<G(NE), 256, 0, stream>>>(raw, stats, s1, nullptr);
    // layer 2: s1 -> s2
    conv64_wmma_kernel<<<dim3(TB * 128), 256, 0, stream>>>(s1, wpck + (size_t)(bi * 3 + 1) * 36864, raw);
    zero_kernel<<<1, 128, 0, stream>>>(stats, 128);
    stats_nhwc64_kernel<<<256, 256, 0, stream>>>(raw, stats);
    bn_finalize_kernel<<<1, 64, 0, stream>>>(stats, F(base + 6), F(base + 7), 64, invTBS);
    bnlif_nhwc_kernel<<<G(NE), 256, 0, stream>>>(raw, stats, s2, nullptr);
    // layer 3: s2 -> (BN + residual cur) -> LIF -> s1
    conv64_wmma_kernel<<<dim3(TB * 128), 256, 0, stream>>>(s2, wpck + (size_t)(bi * 3 + 2) * 36864, raw);
    zero_kernel<<<1, 128, 0, stream>>>(stats, 128);
    stats_nhwc64_kernel<<<256, 256, 0, stream>>>(raw, stats);
    bn_finalize_kernel<<<1, 64, 0, stream>>>(stats, F(base + 10), F(base + 11), 64, invTBS);
    bnlif_nhwc_kernel<<<G(NE), 256, 0, stream>>>(raw, stats, s1, cur);
    bf16_t* ncur = s1; s1 = s2; s2 = cur; cur = ncur;
  }

  // --- outputs: mean_t(out + tmp) then mean_t(est) ---
  final_main_kernel<<<G((long)BATCH * 64 * S_), 256, 0, stream>>>(cur, tmp, (float*)d_out);
  final_est_kernel<<<G((long)BATCH * 4 * S_), 256, 0, stream>>>(est, (float*)d_out + (size_t)BATCH * 64 * S_);
}
